// LSTM_20607253086532
// MI455X (gfx1250) — matrix operand order repaired and checked by reading, not compiled, not run on hardware
//
#include <hip/hip_runtime.h>
#include <stdint.h>

typedef __attribute__((ext_vector_type(16))) __bf16 v16bf;
typedef __attribute__((ext_vector_type(8)))  float  v8f;

#define B_SZ   128
#define T_SZ   1024
#define H_SZ   512
#define KCAT   1024   // I + H
#define NGATE  2048   // 4*H

__device__ __forceinline__ unsigned short f2bf(float f) {
    union { float f; unsigned int u; } v; v.f = f;
    unsigned int r = v.u + 0x7fffu + ((v.u >> 16) & 1u);   // round-to-nearest-even
    return (unsigned short)(r >> 16);
}
__device__ __forceinline__ float sigm(float x) { return 1.0f / (1.0f + __expf(-x)); }
__device__ __forceinline__ float tanh_fast(float x) { return 2.0f / (1.0f + __expf(-2.0f * x)) - 1.0f; }

// Pack W_ih [2048,512] and W_hh [2048,512] into bf16 Wcat [2048,1024] (N-major, K contiguous),
// and fold the two biases into one.
__global__ void pack_weights(const float* __restrict__ w_ih, const float* __restrict__ w_hh,
                             const float* __restrict__ b_ih, const float* __restrict__ b_hh,
                             unsigned short* __restrict__ Wc, float* __restrict__ bias) {
    int n = blockIdx.x;                         // 0..2047 (gate row)
    const float* s0 = w_ih + (size_t)n * H_SZ;
    const float* s1 = w_hh + (size_t)n * H_SZ;
    unsigned short* dst = Wc + (size_t)n * KCAT;
    for (int k = threadIdx.x; k < H_SZ; k += blockDim.x) {
        dst[k]        = f2bf(s0[k]);
        dst[H_SZ + k] = f2bf(s1[k]);
    }
    if (threadIdx.x == 0) bias[n] = b_ih[n] + b_hh[n];
}

// Initialize recurrent-h columns of A buffers (bf16) and cell state (fp32) from h0/c0 [L=2,B,H].
__global__ void init_state(const float* __restrict__ h0, const float* __restrict__ c0,
                           unsigned short* __restrict__ A0, unsigned short* __restrict__ A1,
                           float* __restrict__ C0, float* __restrict__ C1) {
    int idx = blockIdx.x * blockDim.x + threadIdx.x;   // 0..65535
    int b = idx >> 9, j = idx & 511;
    A0[b * KCAT + H_SZ + j] = f2bf(h0[idx]);
    A1[b * KCAT + H_SZ + j] = f2bf(h0[B_SZ * H_SZ + idx]);
    C0[idx] = c0[idx];
    C1[idx] = c0[B_SZ * H_SZ + idx];
}

// Copy x[:, t, :] (fp32, [B,T,I]) into columns 0..511 of A0 as bf16.
__global__ void xcvt(const float* __restrict__ x, unsigned short* __restrict__ A0, int t) {
    int idx = blockIdx.x * blockDim.x + threadIdx.x;   // 0..65535
    int b = idx >> 9, k = idx & 511;
    A0[b * KCAT + k] = f2bf(x[((size_t)b * T_SZ + t) * H_SZ + k]);
}

// Fully fused LSTM cell for one layer, one timestep:
//   gates = A[128,1024](bf16) x Wcat^T + bias   (bf16 WMMA, fp32 acc)
//   c' = sig(f)*c + sig(i)*tanh(g);  h = sig(o)*tanh(c')
// Each wave owns a 16(M) x 16(j) patch and accumulates ALL FOUR gates for it
// (N = j, j+512, j+1024, j+1536), so the pointwise stage runs in-register.
// Grid: 32 blocks (j-slices) x 256 threads (8 waves = 8 M-tiles).
__global__ void lstm_cell(const unsigned short* __restrict__ Au,   // [128][1024] bf16
                          const unsigned short* __restrict__ Wu,   // [2048][1024] bf16
                          const float* __restrict__ bias,          // [2048]
                          float* __restrict__ C,                   // [128][512] cell state
                          unsigned short* __restrict__ h_self,     // A_l + 512 (stride KCAT)
                          unsigned short* __restrict__ h_next,     // A_{l+1} (stride KCAT) or null
                          float* __restrict__ out) {               // d_out or null
    const int wave = threadIdx.x >> 5;
    const int lane = threadIdx.x & 31;
    const int half = lane >> 4;          // 0: lanes 0-15, 1: lanes 16-31
    const int lrow = lane & 15;
    const int j0   = blockIdx.x * 16;    // hidden-column slice 0..511
    const int tm   = wave;               // M tile 0..7

    const unsigned short* arow = Au + (size_t)(tm * 16 + lrow) * KCAT;
    const unsigned short* bi_r = Wu + (size_t)(0 * H_SZ + j0 + lrow) * KCAT;
    const unsigned short* bf_r = Wu + (size_t)(1 * H_SZ + j0 + lrow) * KCAT;
    const unsigned short* bg_r = Wu + (size_t)(2 * H_SZ + j0 + lrow) * KCAT;
    const unsigned short* bo_r = Wu + (size_t)(3 * H_SZ + j0 + lrow) * KCAT;

    v8f acc_i = {0.f,0.f,0.f,0.f,0.f,0.f,0.f,0.f};
    v8f acc_f = acc_i, acc_g = acc_i, acc_o = acc_i;

    union Frag { v16bf v; uint4 q[2]; };

#pragma unroll 2
    for (int k0 = 0; k0 < KCAT; k0 += 32) {
        Frag a, bi, bf, bg, bo;
        // A 16x32 bf16: lane row M=lrow; elems 0-15 -> K = k0 + 16h .. +15 (contiguous)
        a.q[0] = *(const uint4*)(arow + k0 + 16 * half);
        a.q[1] = *(const uint4*)(arow + k0 + 16 * half + 8);
        // B 32x16 bf16: lane col N; elems 0-15 -> K = k0 + 16h .. +15 (contiguous)
        bi.q[0] = *(const uint4*)(bi_r + k0 + 16 * half);
        bi.q[1] = *(const uint4*)(bi_r + k0 + 16 * half + 8);
        bf.q[0] = *(const uint4*)(bf_r + k0 + 16 * half);
        bf.q[1] = *(const uint4*)(bf_r + k0 + 16 * half + 8);
        bg.q[0] = *(const uint4*)(bg_r + k0 + 16 * half);
        bg.q[1] = *(const uint4*)(bg_r + k0 + 16 * half + 8);
        bo.q[0] = *(const uint4*)(bo_r + k0 + 16 * half);
        bo.q[1] = *(const uint4*)(bo_r + k0 + 16 * half + 8);
        acc_i = __builtin_amdgcn_wmma_f32_16x16x32_bf16(false, a.v, false, bi.v, (short)0, acc_i, false, false);
        acc_f = __builtin_amdgcn_wmma_f32_16x16x32_bf16(false, a.v, false, bf.v, (short)0, acc_f, false, false);
        acc_g = __builtin_amdgcn_wmma_f32_16x16x32_bf16(false, a.v, false, bg.v, (short)0, acc_g, false, false);
        acc_o = __builtin_amdgcn_wmma_f32_16x16x32_bf16(false, a.v, false, bo.v, (short)0, acc_o, false, false);
    }

    // In-register pointwise cell update.
    const int   col = j0 + lrow;
    const float bbi = bias[0 * H_SZ + col];
    const float bbf = bias[1 * H_SZ + col];
    const float bbg = bias[2 * H_SZ + col];
    const float bbo = bias[3 * H_SZ + col];

#pragma unroll
    for (int r = 0; r < 8; ++r) {
        const int mrow = tm * 16 + r + 8 * half;   // C/D layout: VGPR r -> M=r (+8 upper half)
        const int idx  = mrow * H_SZ + col;
        float gi = acc_i[r] + bbi;
        float gf = acc_f[r] + bbf;
        float gg = acc_g[r] + bbg;
        float go = acc_o[r] + bbo;
        float c  = sigm(gf) * C[idx] + sigm(gi) * tanh_fast(gg);
        C[idx]   = c;
        float h  = sigm(go) * tanh_fast(c);
        unsigned short hb = f2bf(h);
        h_self[mrow * KCAT + col] = hb;
        if (h_next) h_next[mrow * KCAT + col] = hb;
        if (out)    out[idx] = h;
    }
}

extern "C" void kernel_launch(void* const* d_in, const int* in_sizes, int n_in,
                              void* d_out, int out_size, void* d_ws, size_t ws_size,
                              hipStream_t stream) {
    (void)in_sizes; (void)n_in; (void)out_size; (void)ws_size;
    const float* x     = (const float*)d_in[0];
    const float* h0    = (const float*)d_in[1];
    const float* c0    = (const float*)d_in[2];
    const float* w_ih0 = (const float*)d_in[3];
    const float* w_hh0 = (const float*)d_in[4];
    const float* b_ih0 = (const float*)d_in[5];
    const float* b_hh0 = (const float*)d_in[6];
    const float* w_ih1 = (const float*)d_in[7];
    const float* w_hh1 = (const float*)d_in[8];
    const float* b_ih1 = (const float*)d_in[9];
    const float* b_hh1 = (const float*)d_in[10];
    float* out = (float*)d_out;

    char* ws = (char*)d_ws;
    size_t off = 0;
    auto alloc = [&](size_t bytes) -> char* {
        char* p = ws + off;
        off = (off + bytes + 255) & ~(size_t)255;
        return p;
    };
    unsigned short* A0    = (unsigned short*)alloc((size_t)B_SZ * KCAT * 2);      // 256 KB
    unsigned short* A1    = (unsigned short*)alloc((size_t)B_SZ * KCAT * 2);      // 256 KB
    unsigned short* Wc0   = (unsigned short*)alloc((size_t)NGATE * KCAT * 2);     // 4 MB
    unsigned short* Wc1   = (unsigned short*)alloc((size_t)NGATE * KCAT * 2);     // 4 MB
    float*          bias0 = (float*)alloc((size_t)NGATE * 4);
    float*          bias1 = (float*)alloc((size_t)NGATE * 4);
    float*          C0    = (float*)alloc((size_t)B_SZ * H_SZ * 4);               // 256 KB
    float*          C1    = (float*)alloc((size_t)B_SZ * H_SZ * 4);               // 256 KB

    pack_weights<<<NGATE, 256, 0, stream>>>(w_ih0, w_hh0, b_ih0, b_hh0, Wc0, bias0);
    pack_weights<<<NGATE, 256, 0, stream>>>(w_ih1, w_hh1, b_ih1, b_hh1, Wc1, bias1);
    init_state<<<256, 256, 0, stream>>>(h0, c0, A0, A1, C0, C1);

    for (int t = 0; t < T_SZ; ++t) {
        xcvt<<<256, 256, 0, stream>>>(x, A0, t);
        // Layer 0: h written into A0 recurrent cols and A1 input cols.
        lstm_cell<<<32, 256, 0, stream>>>(A0, Wc0, bias0, C0, A0 + H_SZ, A1, nullptr);
        // Layer 1: h written into A1 recurrent cols; final step also to d_out (fp32).
        lstm_cell<<<32, 256, 0, stream>>>(A1, Wc1, bias1, C1, A1 + H_SZ, nullptr,
                                          (t == T_SZ - 1) ? out : nullptr);
    }
}
